// FUNASR_NANO_ENCODER_19688130085123
// MI455X (gfx1250) — matrix operand order repaired and checked by reading, not compiled, not run on hardware
//
#include <hip/hip_runtime.h>
#include <hip/hip_bf16.h>

// ---- shapes (compile-time) -------------------------------------------------
#define TT   2048
#define DD   512
#define HH   8
#define DKH  64
#define FF   2048
#define KK   11
#define QKV3 1536

typedef __attribute__((ext_vector_type(16))) __bf16 bf16x16;
typedef __attribute__((ext_vector_type(4)))  __bf16 bf16x4;
typedef __attribute__((ext_vector_type(8)))  float  f32x8;

__device__ __forceinline__ f32x8 wmma_bf16(bf16x16 a, bf16x16 b, f32x8 c) {
  // D = A(16x32 bf16) * B(32x16 bf16) + C(16x16 f32)
  return __builtin_amdgcn_wmma_f32_16x16x32_bf16(false, a, false, b, (short)0, c,
                                                 false, false);
}

__device__ __forceinline__ bf16x4 pack4(float4 v) {
  bf16x4 p;
  p[0] = (__bf16)v.x; p[1] = (__bf16)v.y;
  p[2] = (__bf16)v.z; p[3] = (__bf16)v.w;
  return p;
}

// A fragment (16x32, bf16) from row-major LDS tile [rows][ld].
// lane L: m = L&15, half = L>>4; VGPR v<4: k = 8*half+2v ; v>=4: k = 16+8*half+2(v-4)
__device__ __forceinline__ bf16x16 load_A_frag(const __bf16* t, int ld, int mbase,
                                               int kbase, int lane) {
  const __bf16* row = t + (mbase + (lane & 15)) * ld + kbase;
  int h8 = (lane >> 4) * 8;
  bf16x16 a;
#pragma unroll
  for (int v = 0; v < 4; ++v) {
    a[2 * v]     = row[h8 + 2 * v];
    a[2 * v + 1] = row[h8 + 2 * v + 1];
  }
#pragma unroll
  for (int v = 4; v < 8; ++v) {
    a[2 * v]     = row[16 + h8 + 2 * (v - 4)];
    a[2 * v + 1] = row[16 + h8 + 2 * (v - 4) + 1];
  }
  return a;
}

// B fragment (32x16, bf16) from row-major LDS tile [k][n] (ld = row stride).
// lane L: n = L&15, half = L>>4; element i<8: k = 8*half+i ; i>=8: k = 16+8*half+(i-8)
__device__ __forceinline__ bf16x16 load_B_frag(const __bf16* t, int ld, int nbase,
                                               int lane) {
  const __bf16* row = t + nbase + (lane & 15);
  int h8 = (lane >> 4) * 8;
  bf16x16 b;
#pragma unroll
  for (int i = 0; i < 8; ++i) {
    b[i]     = row[(h8 + i) * ld];
    b[8 + i] = row[(16 + h8 + i) * ld];
  }
  return b;
}

// ---- LayerNorm: one wave per 512-element row, float4 vectorized ------------
__global__ void __launch_bounds__(256)
ln_kernel(const float* __restrict__ x, const float* __restrict__ g,
          const float* __restrict__ b, float* __restrict__ out, int rows) {
  int wave = (int)(blockIdx.x * blockDim.x + threadIdx.x) >> 5;
  int lane = threadIdx.x & 31;
  if (wave >= rows) return;
  const float* xr = x + (size_t)wave * DD + lane * 16;
  float4 vals[4];
  float s = 0.f, ss = 0.f;
#pragma unroll
  for (int i = 0; i < 4; ++i) {
    float4 v = *(const float4*)(xr + i * 4);
    vals[i] = v;
    s  += v.x + v.y + v.z + v.w;
    ss += v.x * v.x + v.y * v.y + v.z * v.z + v.w * v.w;
  }
#pragma unroll
  for (int off = 16; off; off >>= 1) {
    s  += __shfl_xor(s,  off, 32);
    ss += __shfl_xor(ss, off, 32);
  }
  float mean = s * (1.0f / DD);
  float var  = ss * (1.0f / DD) - mean * mean;
  float inv  = rsqrtf(var + 1e-5f);
  const float* gp = g + lane * 16;
  const float* bp = b + lane * 16;
  float* orow = out + (size_t)wave * DD + lane * 16;
#pragma unroll
  for (int i = 0; i < 4; ++i) {
    float4 gv = *(const float4*)(gp + i * 4);
    float4 bv = *(const float4*)(bp + i * 4);
    float4 o;
    o.x = (vals[i].x - mean) * inv * gv.x + bv.x;
    o.y = (vals[i].y - mean) * inv * gv.y + bv.y;
    o.z = (vals[i].z - mean) * inv * gv.z + bv.z;
    o.w = (vals[i].w - mean) * inv * gv.w + bv.w;
    *(float4*)(orow + i * 4) = o;
  }
}

// ---- GEMM: C[M,N] = A[M,K] @ W[K,N] + bias (+relu) (+res0) (+res1) ---------
// block = 256 thr (8 waves), tile 128(M) x 128(N); wave tile 32(M) x 64(N).
// Software pipeline: double-buffered LDS, next K-panel global loads issued
// into register temps before the barrier, one barrier per K-step.
__global__ void __launch_bounds__(256)
gemm_kernel(const float* __restrict__ A, const float* __restrict__ W,
            const float* __restrict__ bias,
            const float* __restrict__ res0,
            const float* __restrict__ res1,
            float* __restrict__ C, int M, int Kdim, int N, int relu) {
  __shared__ __bf16 sA[2][128 * 32];   // 2 x 8 KB
  __shared__ __bf16 sB[2][32 * 128];   // 2 x 8 KB
  int tid = threadIdx.x, lane = tid & 31, w = tid >> 5;
  int wm = (w & 3) * 32;            // wave M offset within block tile
  int wn = (w >> 2) * 64;           // wave N offset within block tile
  int mBlk = blockIdx.y * 128, nBlk = blockIdx.x * 128;

  // staging base pointers (advance by constants each K-step)
  const float* aP = A + (size_t)(mBlk + (tid >> 3)) * Kdim + ((tid & 7) << 2);
  const float* bP = W + (size_t)(tid >> 5) * N + nBlk + ((tid & 31) << 2);

  f32x8 acc[2][4];
#pragma unroll
  for (int mi = 0; mi < 2; ++mi)
#pragma unroll
    for (int nt = 0; nt < 4; ++nt) acc[mi][nt] = (f32x8){};

  // prologue: load first K-panel into temps, stage into buffer 0
  float4 ta[4], tb[4];
#pragma unroll
  for (int i = 0; i < 4; ++i) ta[i] = *(const float4*)(aP + (size_t)(32 * i) * Kdim);
#pragma unroll
  for (int i = 0; i < 4; ++i) tb[i] = *(const float4*)(bP + (size_t)(8 * i) * N);
  aP += 32;
  bP += (size_t)32 * N;
  {
    bf16x4* dA = (bf16x4*)sA[0];
    bf16x4* dB = (bf16x4*)sB[0];
#pragma unroll
    for (int i = 0; i < 4; ++i) dA[tid + i * 256] = pack4(ta[i]);
#pragma unroll
    for (int i = 0; i < 4; ++i) dB[tid + i * 256] = pack4(tb[i]);
  }

  int nk = Kdim >> 5;
  int buf = 0;
  for (int kt = 0; kt < nk; ++kt) {
    bool more = (kt + 1 < nk);
    if (more) {                        // issue next panel loads (in flight
#pragma unroll                         //  during this step's WMMAs)
      for (int i = 0; i < 4; ++i) ta[i] = *(const float4*)(aP + (size_t)(32 * i) * Kdim);
#pragma unroll
      for (int i = 0; i < 4; ++i) tb[i] = *(const float4*)(bP + (size_t)(8 * i) * N);
      aP += 32;
      bP += (size_t)32 * N;
    }
    __syncthreads();                   // current buffer fully staged
    const __bf16* cA = sA[buf];
    const __bf16* cB = sB[buf];
    bf16x16 a0 = load_A_frag(cA, 32, wm,      0, lane);
    bf16x16 a1 = load_A_frag(cA, 32, wm + 16, 0, lane);
#pragma unroll
    for (int nt = 0; nt < 4; ++nt) {
      bf16x16 b = load_B_frag(cB, 128, wn + nt * 16, lane);
      acc[0][nt] = wmma_bf16(a0, b, acc[0][nt]);
      acc[1][nt] = wmma_bf16(a1, b, acc[1][nt]);
    }
    if (more) {                        // stage next panel into other buffer
      bf16x4* dA = (bf16x4*)sA[buf ^ 1];
      bf16x4* dB = (bf16x4*)sB[buf ^ 1];
#pragma unroll
      for (int i = 0; i < 4; ++i) dA[tid + i * 256] = pack4(ta[i]);
#pragma unroll
      for (int i = 0; i < 4; ++i) dB[tid + i * 256] = pack4(tb[i]);
      buf ^= 1;
    }
  }

  int half = lane >> 4, col = lane & 15;
#pragma unroll
  for (int mi = 0; mi < 2; ++mi) {
#pragma unroll
    for (int nt = 0; nt < 4; ++nt) {
#pragma unroll
      for (int r = 0; r < 8; ++r) {
        int row = mBlk + wm + mi * 16 + r + 8 * half;
        int cg  = nBlk + wn + nt * 16 + col;
        float v = acc[mi][nt][r] + bias[cg];
        if (relu) v = fmaxf(v, 0.0f);
        if (res0) v += res0[(size_t)row * N + cg];
        if (res1) v += res1[(size_t)row * N + cg];
        C[(size_t)row * N + cg] = v;
      }
    }
  }
}

// ---- FSMN: depthwise conv1d (K=11, pad 5) + identity residual --------------
// 4 channels per thread, float4 loads.
__global__ void __launch_bounds__(256)
fsmn_kernel(const float* __restrict__ qkv,
            const float* __restrict__ fw, float* __restrict__ mem) {
  int i4 = (int)(blockIdx.x * blockDim.x + threadIdx.x) << 2;  // over T*D
  int t = i4 >> 9, d = i4 & (DD - 1);
  const float* base = qkv + 1024 + d;
  float4 acc = *(const float4*)(base + (size_t)t * QKV3);      // + v (residual)
#pragma unroll
  for (int j = 0; j < KK; ++j) {
    int ts = t + j - 5;
    if (ts >= 0 && ts < TT) {
      float4 vv = *(const float4*)(base + (size_t)ts * QKV3);
      acc.x += vv.x * fw[(d + 0) * KK + j];
      acc.y += vv.y * fw[(d + 1) * KK + j];
      acc.z += vv.z * fw[(d + 2) * KK + j];
      acc.w += vv.w * fw[(d + 3) * KK + j];
    }
  }
  *(float4*)(mem + i4) = acc;
}

// ---- Flash attention: block = 1 head x 128 query rows (8 waves x 16) -------
// Double-buffered K/V tiles, one barrier per 32-key step.
__global__ void __launch_bounds__(256)
attn_kernel(const float* __restrict__ qkv, float* __restrict__ attnc) {
  __shared__ __bf16 sKt[2][64 * 32];     // K^T tiles: [dk][key]  (2 x 4 KB)
  __shared__ __bf16 sV[2][32 * 64];      // V tiles:   [key][dk]  (2 x 4 KB)
  __shared__ __bf16 sP[8 * 16 * 32];     // per-wave P tiles      (8 KB)
  int tid = threadIdx.x, lane = tid & 31, w = tid >> 5;
  int h = blockIdx.y;
  int qbase = blockIdx.x * 128 + w * 16;
  int qrow  = qbase + (lane & 15);
  int h8    = (lane >> 4) * 8;
  const float scale = 0.125f;            // dk^-0.5 folded into Q

  // Q fragments (dk 0..31 and 32..63), scaled, loaded once
  bf16x16 qf0, qf1;
  {
    const float* qp = qkv + (size_t)qrow * QKV3 + h * DKH;
#pragma unroll
    for (int v = 0; v < 8; ++v) {
      int k0 = (v < 4) ? (h8 + 2 * v) : (16 + h8 + 2 * (v - 4));
      qf0[2 * v]     = (__bf16)(qp[k0] * scale);
      qf0[2 * v + 1] = (__bf16)(qp[k0 + 1] * scale);
      qf1[2 * v]     = (__bf16)(qp[32 + k0] * scale);
      qf1[2 * v + 1] = (__bf16)(qp[32 + k0 + 1] * scale);
    }
  }

  // staging: thread covers key row (tid>>4)+16i, dk (tid&15)*4
  int kk0 = tid >> 4;
  int d0  = (tid & 15) << 2;
  const float* kP = qkv + (size_t)kk0 * QKV3 + DD + h * DKH + d0;
  const float* vP = kP + DD;

  float mrow[8], lrow[8];
#pragma unroll
  for (int r = 0; r < 8; ++r) { mrow[r] = -1e30f; lrow[r] = 0.0f; }
  f32x8 o0 = {}, o1 = {}, o2 = {}, o3 = {};
  __bf16* sPw = sP + w * 512;

  // prologue: first K/V tile into temps, stage into buffer 0
  float4 tk[2], tv[2];
#pragma unroll
  for (int i = 0; i < 2; ++i) {
    tk[i] = *(const float4*)(kP + (size_t)(16 * i) * QKV3);
    tv[i] = *(const float4*)(vP + (size_t)(16 * i) * QKV3);
  }
  kP += (size_t)32 * QKV3;
  vP += (size_t)32 * QKV3;
#pragma unroll
  for (int i = 0; i < 2; ++i) {
    int kk = kk0 + 16 * i;
    sKt[0][(d0 + 0) * 32 + kk] = (__bf16)tk[i].x;
    sKt[0][(d0 + 1) * 32 + kk] = (__bf16)tk[i].y;
    sKt[0][(d0 + 2) * 32 + kk] = (__bf16)tk[i].z;
    sKt[0][(d0 + 3) * 32 + kk] = (__bf16)tk[i].w;
    *(bf16x4*)(sV[0] + kk * 64 + d0) = pack4(tv[i]);
  }

  int buf = 0;
  for (int step = 0; step < TT / 32; ++step) {
    bool more = (step + 1 < TT / 32);
    if (more) {                          // next tile loads in flight
#pragma unroll
      for (int i = 0; i < 2; ++i) {
        tk[i] = *(const float4*)(kP + (size_t)(16 * i) * QKV3);
        tv[i] = *(const float4*)(vP + (size_t)(16 * i) * QKV3);
      }
      kP += (size_t)32 * QKV3;
      vP += (size_t)32 * QKV3;
    }
    __syncthreads();                     // current buffer staged
    const __bf16* cK = sKt[buf];
    const __bf16* cV = sV[buf];

    // scores S[16 x 32] = Q(16x64) . K^T(64x32)
    f32x8 s0t = {}, s1t = {};
    s0t = wmma_bf16(qf0, load_B_frag(cK,           32, 0,  lane), s0t);
    s0t = wmma_bf16(qf1, load_B_frag(cK + 32 * 32, 32, 0,  lane), s0t);
    s1t = wmma_bf16(qf0, load_B_frag(cK,           32, 16, lane), s1t);
    s1t = wmma_bf16(qf1, load_B_frag(cK + 32 * 32, 32, 16, lane), s1t);

    // online softmax (rows striped: VGPR r across a 16-lane half-group)
#pragma unroll
    for (int r = 0; r < 8; ++r) {
      float mt = fmaxf(s0t[r], s1t[r]);
#pragma unroll
      for (int off = 8; off; off >>= 1) mt = fmaxf(mt, __shfl_xor(mt, off, 16));
      float mn = fmaxf(mrow[r], mt);
      float al = expf(mrow[r] - mn);
      mrow[r] = mn;
      float p0 = expf(s0t[r] - mn);
      float p1 = expf(s1t[r] - mn);
      float ps = p0 + p1;
#pragma unroll
      for (int off = 8; off; off >>= 1) ps += __shfl_xor(ps, off, 16);
      lrow[r] = lrow[r] * al + ps;
      int rl = r + h8;                   // local row 0..15
      sPw[rl * 32 + (lane & 15)]      = (__bf16)p0;
      sPw[rl * 32 + 16 + (lane & 15)] = (__bf16)p1;
      o0[r] = o0[r] * al; o1[r] = o1[r] * al;
      o2[r] = o2[r] * al; o3[r] = o3[r] * al;
    }

    // O += P(16x32) . V(32x64)   (sP write->read is wave-local, LDS in-order)
    bf16x16 pa = load_A_frag(sPw, 32, 0, 0, lane);
    o0 = wmma_bf16(pa, load_B_frag(cV, 64, 0,  lane), o0);
    o1 = wmma_bf16(pa, load_B_frag(cV, 64, 16, lane), o1);
    o2 = wmma_bf16(pa, load_B_frag(cV, 64, 32, lane), o2);
    o3 = wmma_bf16(pa, load_B_frag(cV, 64, 48, lane), o3);

    if (more) {                          // stage next tile into other buffer
      __bf16* nK = sKt[buf ^ 1];
      __bf16* nV = sV[buf ^ 1];
#pragma unroll
      for (int i = 0; i < 2; ++i) {
        int kk = kk0 + 16 * i;
        nK[(d0 + 0) * 32 + kk] = (__bf16)tk[i].x;
        nK[(d0 + 1) * 32 + kk] = (__bf16)tk[i].y;
        nK[(d0 + 2) * 32 + kk] = (__bf16)tk[i].z;
        nK[(d0 + 3) * 32 + kk] = (__bf16)tk[i].w;
        *(bf16x4*)(nV + kk * 64 + d0) = pack4(tv[i]);
      }
      buf ^= 1;
    }
  }

  int col = lane & 15;
#pragma unroll
  for (int r = 0; r < 8; ++r) {
    float inv = 1.0f / lrow[r];
    int t = qbase + r + h8;
    float* orow = attnc + (size_t)t * DD + h * DKH;
    orow[col]      = o0[r] * inv;
    orow[16 + col] = o1[r] * inv;
    orow[32 + col] = o2[r] * inv;
    orow[48 + col] = o3[r] * inv;
  }
}

// ---- host orchestration ----------------------------------------------------
extern "C" void kernel_launch(void* const* d_in, const int* in_sizes, int n_in,
                              void* d_out, int out_size, void* d_ws, size_t ws_size,
                              hipStream_t stream) {
  (void)in_sizes; (void)n_in; (void)out_size; (void)ws_size;
  const float* x_in    = (const float*)d_in[0];
  const float* ln1_g   = (const float*)d_in[1];
  const float* ln1_b   = (const float*)d_in[2];
  const float* Wqkv    = (const float*)d_in[3];
  const float* bqkv    = (const float*)d_in[4];
  const float* fsmn_w  = (const float*)d_in[5];
  const float* Wout    = (const float*)d_in[6];
  const float* bout    = (const float*)d_in[7];
  const float* ln2_g   = (const float*)d_in[8];
  const float* ln2_b   = (const float*)d_in[9];
  const float* W1      = (const float*)d_in[10];
  const float* b1      = (const float*)d_in[11];
  const float* W2      = (const float*)d_in[12];
  const float* b2      = (const float*)d_in[13];
  const float* after_g = (const float*)d_in[14];
  const float* after_b = (const float*)d_in[15];

  const size_t TD = (size_t)TT * DD;      // 1M floats
  float* ws    = (float*)d_ws;
  float* xA    = ws;                      // post-attention x
  float* xB    = ws + 1 * TD;             // post-FFN x (layer output)
  float* x1    = ws + 2 * TD;             // LN output
  float* qkvb  = ws + 3 * TD;             // [T, 3D] (3M floats)
  float* memb  = ws + 6 * TD;             // FSMN out
  float* attnc = ws + 7 * TD;             // attention context
  float* hbuf  = ws + 8 * TD;             // [T, F]  (4M floats)

  const float* xcur = x_in;
  for (int l = 0; l < 8; ++l) {
    ln_kernel<<<256, 256, 0, stream>>>(xcur, ln1_g + l * DD, ln1_b + l * DD, x1, TT);
    gemm_kernel<<<dim3(QKV3 / 128, TT / 128), 256, 0, stream>>>(
        x1, Wqkv + (size_t)l * DD * QKV3, bqkv + (size_t)l * QKV3,
        nullptr, nullptr, qkvb, TT, DD, QKV3, 0);
    fsmn_kernel<<<(TT * DD) / 1024, 256, 0, stream>>>(qkvb, fsmn_w + (size_t)l * DD * KK, memb);
    attn_kernel<<<dim3(TT / 128, HH), 256, 0, stream>>>(qkvb, attnc);
    gemm_kernel<<<dim3(DD / 128, TT / 128), 256, 0, stream>>>(
        attnc, Wout + (size_t)l * DD * DD, bout + (size_t)l * DD,
        xcur, memb, xA, TT, DD, DD, 0);
    ln_kernel<<<256, 256, 0, stream>>>(xA, ln2_g + l * DD, ln2_b + l * DD, x1, TT);
    gemm_kernel<<<dim3(FF / 128, TT / 128), 256, 0, stream>>>(
        x1, W1 + (size_t)l * DD * FF, b1 + (size_t)l * FF,
        nullptr, nullptr, hbuf, TT, DD, FF, 1);
    gemm_kernel<<<dim3(DD / 128, TT / 128), 256, 0, stream>>>(
        hbuf, W2 + (size_t)l * FF * DD, b2 + (size_t)l * DD,
        xA, nullptr, xB, TT, FF, DD, 0);
    xcur = xB;
  }
  ln_kernel<<<256, 256, 0, stream>>>(xcur, after_g, after_b, (float*)d_out, TT);
}
